// Transformer_62036507623879
// MI455X (gfx1250) — matrix operand order repaired and checked by reading, not compiled, not run on hardware
//
#include <hip/hip_runtime.h>
#include <hip/hip_bf16.h>
#include <stdint.h>

// ---------------------------------------------------------------------------
// Problem constants (match reference)
// ---------------------------------------------------------------------------
#define BB  4
#define SS  512
#define DD  1024
#define HH  16
#define LL  4
#define FF  4096
#define VV  8000
#define DHH 64
#define MTOK (BB * SS)          // 2048 token rows
#define EPSV 1e-5f

// GEMM tiling
#define BM 128
#define BN 128
#define BK 32
#define PADH 2                  // half-element pad -> 17-dword skew, conflict free
#define PADC 4                  // f32 pad for the C staging buffer

typedef __attribute__((ext_vector_type(16))) _Float16 v16h;
typedef __attribute__((ext_vector_type(2)))  _Float16 h2v;
typedef __attribute__((ext_vector_type(8)))  float    v8f;

union Frag16 { v16h v; uint32_t u[8]; };

// ---------------------------------------------------------------------------
// Generic batched GEMM:  C = scale * (A @ B) + bias   (optional relu)
//   A: [M,K] f32, row stride lda, batch offset zb*aSB + zh*aSH
//   B: BTRANS==0 : [K,N] f32, element B[k*ldb+n]
//      BTRANS==1 : [N,K] f32, element B[n*ldb+k]
//   C: [M,N] f32, row stride ldc
//   z = blockIdx.z;  zb = z / Hdiv;  zh = z % Hdiv
// Requirements (all call sites satisfy): M % 128 == 0, K % 32 == 0,
// N % 16 == 0 (N % 4 == 0 for partial tiles), pointers 16B aligned,
// lda/ldb/ldc % 4 == 0.
// Internally converts f32->f16 and uses v_wmma_f32_16x16x32_f16.
// ---------------------------------------------------------------------------
struct GemmArgs {
    const float* A;
    const float* Bm;
    const float* bias;          // may be null
    float*       C;
    int M, N, K;
    int lda, ldb, ldc;
    long long aSB, aSH, bSB, bSH, cSB, cSH;
    int Hdiv;
    float scale;
    int relu;
};

// LDS: f16 staging tiles for the K-loop, aliased with the f32 C-staging
// buffer used by the coalescing epilogue (union keeps LDS at ~33 KB).
union SmemU {
    struct { _Float16 A[BM][BK + PADH]; _Float16 B[BN][BK + PADH]; } st;
    float cst[64][BN + PADC];
};

template<int BTRANS>
__global__ __launch_bounds__(256) void gemm_wmma_f16(GemmArgs p) {
    __shared__ SmemU sm;

    const int tid  = threadIdx.x;
    const int lane = tid & 31;
    const int w    = tid >> 5;       // 8 waves
    const int wm   = w >> 1;         // 0..3  -> 32 rows each
    const int wn   = w & 1;          // 0..1  -> 64 cols each

    const int z  = blockIdx.z;
    const int zb = z / p.Hdiv;
    const int zh = z % p.Hdiv;
    const float* __restrict__ Ab = p.A  + zb * p.aSB + zh * p.aSH;
    const float* __restrict__ Bb = p.Bm + zb * p.bSB + zh * p.bSH;
    float*       __restrict__ Cb = p.C  + zb * p.cSB + zh * p.cSH;

    const int m0 = blockIdx.y * BM;
    const int n0 = blockIdx.x * BN;
    const bool fullN = (n0 + BN <= p.N);

    v8f acc[2][4] = {};

    // register staging for the software pipeline (4 quads each = 16 floats)
    float4 aReg[4];
    float4 bReg[4];

    auto loadA = [&](int kt) {
        #pragma unroll
        for (int it = 0; it < 4; ++it) {
            const int i  = tid + it * 256;   // quad index
            const int r  = i >> 3;           // 8 quads per row (BK/4)
            const int kq = (i & 7) * 4;
            aReg[it] = *(const float4*)&Ab[(long long)(m0 + r) * p.lda + kt + kq];
        }
    };
    auto loadB = [&](int kt) {
        #pragma unroll
        for (int it = 0; it < 4; ++it) {
            const int i = tid + it * 256;
            if (BTRANS) {                       // B[n,k]: read 4 consecutive k
                const int c  = i >> 3;
                const int kq = (i & 7) * 4;
                const int gc = n0 + c;
                if (fullN || gc < p.N)
                    bReg[it] = *(const float4*)&Bb[(long long)gc * p.ldb + kt + kq];
                else
                    bReg[it] = make_float4(0.f, 0.f, 0.f, 0.f);
            } else {                            // B[k,n]: read 4 consecutive n
                const int kk = i >> 5;          // 32 quads per k-row (BN/4)
                const int cq = (i & 31) * 4;
                const int gc = n0 + cq;
                if (fullN || gc + 4 <= p.N)     // N%4==0 -> quad all-in or all-out
                    bReg[it] = *(const float4*)&Bb[(long long)(kt + kk) * p.ldb + gc];
                else
                    bReg[it] = make_float4(0.f, 0.f, 0.f, 0.f);
            }
        }
    };
    auto storeA = [&]() {
        #pragma unroll
        for (int it = 0; it < 4; ++it) {
            const int i  = tid + it * 256;
            const int r  = i >> 3;
            const int kq = (i & 7) * 4;
            h2v p0 = { (_Float16)aReg[it].x, (_Float16)aReg[it].y };
            h2v p1 = { (_Float16)aReg[it].z, (_Float16)aReg[it].w };
            *(h2v*)&sm.st.A[r][kq]     = p0;
            *(h2v*)&sm.st.A[r][kq + 2] = p1;
        }
    };
    auto storeB = [&]() {
        #pragma unroll
        for (int it = 0; it < 4; ++it) {
            const int i = tid + it * 256;
            if (BTRANS) {
                const int c  = i >> 3;
                const int kq = (i & 7) * 4;
                h2v p0 = { (_Float16)bReg[it].x, (_Float16)bReg[it].y };
                h2v p1 = { (_Float16)bReg[it].z, (_Float16)bReg[it].w };
                *(h2v*)&sm.st.B[c][kq]     = p0;
                *(h2v*)&sm.st.B[c][kq + 2] = p1;
            } else {
                const int kk = i >> 5;
                const int cq = (i & 31) * 4;
                sm.st.B[cq + 0][kk] = (_Float16)bReg[it].x;
                sm.st.B[cq + 1][kk] = (_Float16)bReg[it].y;
                sm.st.B[cq + 2][kk] = (_Float16)bReg[it].z;
                sm.st.B[cq + 3][kk] = (_Float16)bReg[it].w;
            }
        }
    };

    // prologue: stage first tile
    loadA(0); loadB(0);
    storeA(); storeB();
    __syncthreads();

    for (int kt = 0; kt < p.K; kt += BK) {
        const bool haveNext = (kt + BK < p.K);
        if (haveNext) { loadA(kt + BK); loadB(kt + BK); }   // overlap with WMMA

        // ---- fragments in contiguous k order (k = 16*(lane/16) + e), same for A and B -----------------------
        Frag16 afr[2];
        {
            const int hb = (lane >> 4) * 16;         // lanes 16-31 hold K=16..31
            #pragma unroll
            for (int mt = 0; mt < 2; ++mt) {
                const int r = wm * 32 + mt * 16 + (lane & 15);
                #pragma unroll
                for (int s = 0; s < 8; ++s) {
                    const int kk = s * 2 + hb;
                    afr[mt].u[s] = *(const uint32_t*)&sm.st.A[r][kk];
                }
            }
        }
        Frag16 bfr[4];
        {
            const int kb = (lane >> 4) * 16;         // lanes 16-31 hold K=16..31
            #pragma unroll
            for (int nt = 0; nt < 4; ++nt) {
                const int c = wn * 64 + nt * 16 + (lane & 15);
                #pragma unroll
                for (int s = 0; s < 8; ++s)
                    bfr[nt].u[s] = *(const uint32_t*)&sm.st.B[c][kb + s * 2];
            }
        }

        // ---- 8x v_wmma_f32_16x16x32_f16 ------------------------------------
        #pragma unroll
        for (int mt = 0; mt < 2; ++mt)
            #pragma unroll
            for (int nt = 0; nt < 4; ++nt)
                acc[mt][nt] = __builtin_amdgcn_wmma_f32_16x16x32_f16(
                    false, afr[mt].v, false, bfr[nt].v,
                    (short)0, acc[mt][nt], false, false);

        __syncthreads();
        if (haveNext) {
            storeA(); storeB();
            __syncthreads();
        }
    }

    // ---- epilogue: stage through LDS, then coalesced b128 stores ----------
    // C/D VGPR layout: VGPR i -> row (lane<16 ? i : 8+i), col = lane&15.
    const int rsel = (lane >> 4) * 8;
    #pragma unroll
    for (int mt = 0; mt < 2; ++mt) {
        __syncthreads();                      // LDS reuse: frag reads / prev pass done
        const int rslot0 = wm * 16 + rsel;    // 64 distinct rows per mt pass
        #pragma unroll
        for (int nt = 0; nt < 4; ++nt) {
            const int c = wn * 64 + nt * 16 + (lane & 15);
            #pragma unroll
            for (int i = 0; i < 8; ++i)
                sm.cst[rslot0 + i][c] = acc[mt][nt][i];
        }
        __syncthreads();
        // 64 rows x 128 cols = 2048 quads, 8 per thread, coalesced rows
        #pragma unroll
        for (int it = 0; it < 8; ++it) {
            const int qi   = tid + it * 256;
            const int slot = qi >> 5;              // 32 quads per row
            const int cq   = (qi & 31) * 4;
            const int gcol = n0 + cq;
            if (fullN || gcol + 4 <= p.N) {        // N%4==0 -> quad all-in/out
                const int grow = m0 + (slot >> 4) * 32 + mt * 16 + (slot & 15);
                float4 vv;
                vv.x = sm.cst[slot][cq + 0] * p.scale;
                vv.y = sm.cst[slot][cq + 1] * p.scale;
                vv.z = sm.cst[slot][cq + 2] * p.scale;
                vv.w = sm.cst[slot][cq + 3] * p.scale;
                if (p.bias) {
                    const float4 b4 = *(const float4*)&p.bias[gcol];
                    vv.x += b4.x; vv.y += b4.y; vv.z += b4.z; vv.w += b4.w;
                }
                if (p.relu) {
                    vv.x = fmaxf(vv.x, 0.f); vv.y = fmaxf(vv.y, 0.f);
                    vv.z = fmaxf(vv.z, 0.f); vv.w = fmaxf(vv.w, 0.f);
                }
                *(float4*)&Cb[(long long)grow * p.ldc + gcol] = vv;
            }
        }
    }
}

// ---------------------------------------------------------------------------
// Embedding gather + (buggy-faithful all-sin) positional encoding
// one block per token; writes x (ws) and target_embeddings (d_out tail)
// ---------------------------------------------------------------------------
__global__ __launch_bounds__(256) void embed_pe_kernel(const int* __restrict__ ids,
                                                       const float* __restrict__ emb,
                                                       float* __restrict__ x,
                                                       float* __restrict__ temb) {
    const int token = blockIdx.x;          // b*S + s
    const int spos  = token & (SS - 1);
    const int id    = ids[token];
    const float pos = (float)spos;
    #pragma unroll
    for (int jj = 0; jj < DD / 256; ++jj) {
        const int j = threadIdx.x + jj * 256;
        float ex;
        if (j < DD / 2) ex = (2.0f * (float)j) / (float)DD;                   // freqs[::2]
        else            ex = (2.0f * (float)(j - DD / 2) + 1.0f) / (float)DD; // freqs[1::2]
        const float pe = __sinf(pos * __powf(10000.0f, -ex));
        const float vv = emb[(long long)id * DD + j] + pe;
        x[(long long)token * DD + j]    = vv;
        temb[(long long)token * DD + j] = vv;
    }
}

// ---------------------------------------------------------------------------
// Masked softmax over one attention row (512 cols), in place.
// rowid = ((b*H + h)*S + q);  block = 256 threads, 2 cols each.
// ---------------------------------------------------------------------------
__global__ __launch_bounds__(256) void softmax_mask_kernel(float* __restrict__ attn,
                                                           const int* __restrict__ lens) {
    const int rowid = blockIdx.x;
    const int qpos  = rowid & (SS - 1);
    const int b     = rowid >> 13;               // / (H*S)
    float* row = attn + (long long)rowid * SS;
    const int len  = lens[b];
    const bool qv  = qpos < len;
    const int tid  = threadIdx.x;

    float vloc[2]; bool mloc[2];
    float mx = -3.0e38f;
    #pragma unroll
    for (int j = 0; j < 2; ++j) {
        const int c = tid + j * 256;
        const bool m = qv && (c < len) && (c <= qpos);
        const float s = row[c];
        const float vv = m ? s : -1.0e9f;
        mloc[j] = m; vloc[j] = vv;
        mx = fmaxf(mx, vv);
    }
    __shared__ float redA[8], redB[8];
    #pragma unroll
    for (int o = 16; o > 0; o >>= 1) mx = fmaxf(mx, __shfl_xor(mx, o, 32));
    if ((tid & 31) == 0) redA[tid >> 5] = mx;
    __syncthreads();
    float bmax = redA[0];
    #pragma unroll
    for (int i = 1; i < 8; ++i) bmax = fmaxf(bmax, redA[i]);

    float e[2]; float sum = 0.0f;
    #pragma unroll
    for (int j = 0; j < 2; ++j) { e[j] = __expf(vloc[j] - bmax); sum += e[j]; }
    #pragma unroll
    for (int o = 16; o > 0; o >>= 1) sum += __shfl_xor(sum, o, 32);
    if ((tid & 31) == 0) redB[tid >> 5] = sum;
    __syncthreads();
    float bsum = 0.0f;
    #pragma unroll
    for (int i = 0; i < 8; ++i) bsum += redB[i];
    const float inv = 1.0f / bsum;

    #pragma unroll
    for (int j = 0; j < 2; ++j) {
        const int c = tid + j * 256;
        row[c] = mloc[j] ? e[j] * inv : 0.0f;
    }
}

// ---------------------------------------------------------------------------
// x = LayerNorm(x + t) * g + be   (row = token, D = 1024, 4 elems/thread)
// ---------------------------------------------------------------------------
__global__ __launch_bounds__(256) void add_ln_kernel(float* __restrict__ x,
                                                     const float* __restrict__ t,
                                                     const float* __restrict__ g,
                                                     const float* __restrict__ be) {
    const int row = blockIdx.x;
    float* xr = x + (long long)row * DD;
    const float* tr = t + (long long)row * DD;
    const int tid = threadIdx.x;

    float vals[4]; float s = 0.0f;
    #pragma unroll
    for (int j = 0; j < 4; ++j) {
        const int c = tid + j * 256;
        vals[j] = xr[c] + tr[c];
        s += vals[j];
    }
    __shared__ float redA[8], redB[8];
    #pragma unroll
    for (int o = 16; o > 0; o >>= 1) s += __shfl_xor(s, o, 32);
    if ((tid & 31) == 0) redA[tid >> 5] = s;
    __syncthreads();
    float tot = 0.0f;
    #pragma unroll
    for (int i = 0; i < 8; ++i) tot += redA[i];
    const float mean = tot * (1.0f / (float)DD);

    float vs = 0.0f;
    #pragma unroll
    for (int j = 0; j < 4; ++j) { const float d = vals[j] - mean; vs += d * d; }
    #pragma unroll
    for (int o = 16; o > 0; o >>= 1) vs += __shfl_xor(vs, o, 32);
    if ((tid & 31) == 0) redB[tid >> 5] = vs;
    __syncthreads();
    float vtot = 0.0f;
    #pragma unroll
    for (int i = 0; i < 8; ++i) vtot += redB[i];
    const float inv = rsqrtf(vtot * (1.0f / (float)DD) + EPSV);

    #pragma unroll
    for (int j = 0; j < 4; ++j) {
        const int c = tid + j * 256;
        xr[c] = (vals[j] - mean) * inv * g[c] + be[c];
    }
}

// ---------------------------------------------------------------------------
// Host-side GEMM launcher
// ---------------------------------------------------------------------------
static inline void launch_gemm(hipStream_t stream,
                               const float* A, const float* Bm, const float* bias, float* C,
                               int M, int N, int K, int lda, int ldb, int ldc,
                               long long aSB, long long aSH,
                               long long bSB, long long bSH,
                               long long cSB, long long cSH,
                               int Hdiv, int batch,
                               float scale, int btrans, int relu) {
    GemmArgs p;
    p.A = A; p.Bm = Bm; p.bias = bias; p.C = C;
    p.M = M; p.N = N; p.K = K;
    p.lda = lda; p.ldb = ldb; p.ldc = ldc;
    p.aSB = aSB; p.aSH = aSH; p.bSB = bSB; p.bSH = bSH; p.cSB = cSB; p.cSH = cSH;
    p.Hdiv = Hdiv; p.scale = scale; p.relu = relu;
    dim3 grid((N + BN - 1) / BN, (M + BM - 1) / BM, batch);
    if (btrans)
        gemm_wmma_f16<1><<<grid, dim3(256), 0, stream>>>(p);
    else
        gemm_wmma_f16<0><<<grid, dim3(256), 0, stream>>>(p);
}

// ---------------------------------------------------------------------------
// kernel_launch
// ---------------------------------------------------------------------------
extern "C" void kernel_launch(void* const* d_in, const int* in_sizes, int n_in,
                              void* d_out, int out_size, void* d_ws, size_t ws_size,
                              hipStream_t stream) {
    const int*   ids  = (const int*)  d_in[0];
    const int*   lens = (const int*)  d_in[1];
    const float* emb  = (const float*)d_in[2];
    const float* Wq   = (const float*)d_in[3];
    const float* bq   = (const float*)d_in[4];
    const float* Wk   = (const float*)d_in[5];
    const float* bk   = (const float*)d_in[6];
    const float* Wv   = (const float*)d_in[7];
    const float* bv   = (const float*)d_in[8];
    const float* Wo   = (const float*)d_in[9];
    const float* bo   = (const float*)d_in[10];
    const float* W1   = (const float*)d_in[11];
    const float* b1   = (const float*)d_in[12];
    const float* W2   = (const float*)d_in[13];
    const float* b2   = (const float*)d_in[14];
    const float* g1   = (const float*)d_in[15];
    const float* be1  = (const float*)d_in[16];
    const float* g2   = (const float*)d_in[17];
    const float* be2  = (const float*)d_in[18];
    const float* Wc   = (const float*)d_in[19];
    const float* bc   = (const float*)d_in[20];
    (void)n_in; (void)in_sizes; (void)out_size; (void)ws_size;

    // d_out partition: [logits | aligns | target_embeddings]
    float* logits = (float*)d_out;
    float* aligns = logits + (long long)BB * SS * VV;
    float* temb   = aligns + (long long)LL * BB * HH * SS * SS;

    // workspace partition (f32)
    const long long TD = (long long)MTOK * DD;   // 2,097,152
    float* x  = (float*)d_ws;
    float* q  = x + TD;
    float* k  = q + TD;
    float* v  = k + TD;
    float* y  = v + TD;
    float* t  = y + TD;
    float* ff = t + TD;                           // MTOK * FF

    // 1) embedding + positional encoding (+ target_embeddings output)
    embed_pe_kernel<<<dim3(MTOK), dim3(256), 0, stream>>>(ids, emb, x, temb);

    const long long sSD  = (long long)SS * DD;        // per-batch stride in q/k/v/y
    const long long sHSS = (long long)HH * SS * SS;   // per-batch stride in attn
    const long long sSS  = (long long)SS * SS;        // per-head stride in attn

    for (int l = 0; l < LL; ++l) {
        const float* Wql = Wq + (long long)l * DD * DD;
        const float* Wkl = Wk + (long long)l * DD * DD;
        const float* Wvl = Wv + (long long)l * DD * DD;
        const float* Wol = Wo + (long long)l * DD * DD;
        const float* W1l = W1 + (long long)l * DD * FF;
        const float* W2l = W2 + (long long)l * FF * DD;
        float* attn_l = aligns + (long long)l * BB * HH * SS * SS;

        // q/k/v projections: [2048,1024] = x @ W + b
        launch_gemm(stream, x, Wql, bq + l * DD, q, MTOK, DD, DD, DD, DD, DD,
                    0, 0, 0, 0, 0, 0, 1, 1, 1.0f, 0, 0);
        launch_gemm(stream, x, Wkl, bk + l * DD, k, MTOK, DD, DD, DD, DD, DD,
                    0, 0, 0, 0, 0, 0, 1, 1, 1.0f, 0, 0);
        launch_gemm(stream, x, Wvl, bv + l * DD, v, MTOK, DD, DD, DD, DD, DD,
                    0, 0, 0, 0, 0, 0, 1, 1, 1.0f, 0, 0);

        // scores[b,h] = (Q_bh @ K_bh^T) / sqrt(dh)   -> written into d_out aligns
        launch_gemm(stream, q, k, nullptr, attn_l,
                    SS, SS, DHH, DD, DD, SS,
                    sSD, DHH, sSD, DHH, sHSS, sSS,
                    HH, BB * HH, 0.125f, /*btrans=*/1, 0);

        // masked softmax in place (one block per attention row)
        softmax_mask_kernel<<<dim3(BB * HH * SS), dim3(256), 0, stream>>>(attn_l, lens);

        // y[b,h] = attn_bh @ V_bh   (N=64 per head, written into [B,S,D] layout)
        launch_gemm(stream, attn_l, v, nullptr, y,
                    SS, DHH, SS, SS, DD, DD,
                    sHSS, sSS, sSD, DHH, sSD, DHH,
                    HH, BB * HH, 1.0f, 0, 0);

        // t = y @ Wo + bo ; x = LN(x + t)
        launch_gemm(stream, y, Wol, bo + l * DD, t, MTOK, DD, DD, DD, DD, DD,
                    0, 0, 0, 0, 0, 0, 1, 1, 1.0f, 0, 0);
        add_ln_kernel<<<dim3(MTOK), dim3(256), 0, stream>>>(x, t, g1 + l * DD, be1 + l * DD);

        // ff = relu(x @ W1 + b1) ; t = ff @ W2 + b2 ; x = LN(x + t)
        launch_gemm(stream, x, W1l, b1 + l * FF, ff, MTOK, FF, DD, DD, FF, FF,
                    0, 0, 0, 0, 0, 0, 1, 1, 1.0f, 0, /*relu=*/1);
        launch_gemm(stream, ff, W2l, b2 + l * DD, t, MTOK, DD, FF, FF, DD, DD,
                    0, 0, 0, 0, 0, 0, 1, 1, 1.0f, 0, 0);
        add_ln_kernel<<<dim3(MTOK), dim3(256), 0, stream>>>(x, t, g2 + l * DD, be2 + l * DD);
    }

    // classifier: logits = x @ Wc + bc   (N = 8000, guarded tiles)
    launch_gemm(stream, x, Wc, bc, logits, MTOK, VV, DD, DD, VV, VV,
                0, 0, 0, 0, 0, 0, 1, 1, 1.0f, 0, 0);
}
